// BasicGraphModel_51333449121924
// MI455X (gfx1250) — matrix operand order repaired and checked by reading, not compiled, not run on hardware
//
#include <hip/hip_runtime.h>
#include <hip/hip_bf16.h>

// ---------------------------------------------------------------------------
// GCN (3x GCNConv + ELU) for MI455X / gfx1250.
//  - GEMMs via v_wmma_f32_16x16x32_bf16; B (weights) register-resident,
//    A streamed; wave32, 16x16 tiles, f32 accumulators.
//  - Edge aggregation: one wave per edge, hw f32 atomics (L2-resident).
// ---------------------------------------------------------------------------

#define NNODES 50000
#define NEDGES 800000
#define NPAD   50176        // 196 * 256

typedef __attribute__((ext_vector_type(16))) __bf16 v16bf;
typedef __attribute__((ext_vector_type(8)))  __bf16 v8bf;
typedef __attribute__((ext_vector_type(8)))  float  v8f;
typedef __attribute__((ext_vector_type(4)))  float  v4f;

union ABfrag { v16bf v; v8bf h[2]; };

// ------------------------- degree / normalization --------------------------
__global__ void k_deg_init(float* __restrict__ deg, int n) {
  int i = blockIdx.x * blockDim.x + threadIdx.x;
  if (i < n) deg[i] = 1.0f;                       // self-loop contribution
}

__global__ void k_deg_count(const int* __restrict__ dst, float* __restrict__ deg, int e) {
  int i = blockIdx.x * blockDim.x + threadIdx.x;
  if (i < e) unsafeAtomicAdd(&deg[dst[i]], 1.0f);
}

__global__ void k_deg_rsqrt(float* __restrict__ deg, int n) {
  int i = blockIdx.x * blockDim.x + threadIdx.x;
  if (i < n) deg[i] = rsqrtf(deg[i]);             // deg >= 1 always
}

// ----------------------------- staging ------------------------------------
// f32 -> bf16 with zero padding of rows [NNODES, NPAD)
__global__ void k_f32_to_bf16_pad(const float* __restrict__ src, __bf16* __restrict__ dst,
                                  int nvalid, int ntotal) {
  int i = blockIdx.x * blockDim.x + threadIdx.x;
  if (i < ntotal) dst[i] = (i < nvalid) ? (__bf16)src[i] : (__bf16)0.0f;
}

// W [128 x ncols] row-major f32  ->  Wt [ncols x 128] (column-major W) bf16
__global__ void k_wt_bf16(const float* __restrict__ W, __bf16* __restrict__ Wt, int ncols) {
  int i = blockIdx.x * blockDim.x + threadIdx.x;
  if (i < 128 * ncols) {
    int n = i >> 7;          // output column
    int k = i & 127;         // K index
    Wt[i] = (__bf16)W[k * ncols + n];
  }
}

// ------------------------------- GEMM --------------------------------------
// Xb: [NPAD x 128] bf16 row-major.  Wt: [NT*16 x 128] bf16 (W columns contiguous).
// Hout: [NPAD x NT*16] f32.
// Block = 256 threads = 8 waves, covering 256 rows x all NT*16 columns.
// Wave = one column-pair (32 cols) x (GROUPS*16) rows; its 8 B fragments live
// in VGPRs for the whole kernel, only A is streamed.
template <int NT>
__global__ __launch_bounds__(256)
void k_gemm_bf16(const __bf16* __restrict__ Xb, const __bf16* __restrict__ Wt,
                 float* __restrict__ Hout) {
  constexpr int NCOLS  = NT * 16;
  constexpr int CP     = NT / 2;        // column pairs across waves (4 or 2)
  constexpr int WPC    = 8 / CP;        // row groups across waves   (2 or 4)
  constexpr int GROUPS = 16 / WPC;      // 16-row groups per wave    (8 or 4)

  const int wave  = threadIdx.x >> 5;
  const int lane  = threadIdx.x & 31;
  const int m     = lane & 15;
  const int hi    = lane >> 4;          // lane half selects K sub-chunk
  const int cpair = wave % CP;          // which 32-column pair
  const int rgrp  = wave / CP;          // which row strip
  const int rowbase0 = blockIdx.x * 256 + rgrp * (GROUPS * 16);

  // ---- load B fragments once: 2 tiles x 4 k-steps, kept in registers ----
  ABfrag B[4][2];
#pragma unroll
  for (int kt = 0; kt < 4; ++kt)
#pragma unroll
    for (int t = 0; t < 2; ++t) {
      const __bf16* bcol = Wt + (size_t)(cpair * 32 + t * 16 + m) * 128 + kt * 32 + hi * 16;
      B[kt][t].h[0] = *(const v8bf*)(bcol);
      B[kt][t].h[1] = *(const v8bf*)(bcol + 8);
    }

  // ---- march down rows; stream A, accumulate in registers ----
  for (int g = 0; g < GROUPS; ++g) {
    const int rowbase = rowbase0 + g * 16;
    const __bf16* arow = Xb + (size_t)(rowbase + m) * 128;
    v8f acc0 = {0.f, 0.f, 0.f, 0.f, 0.f, 0.f, 0.f, 0.f};
    v8f acc1 = {0.f, 0.f, 0.f, 0.f, 0.f, 0.f, 0.f, 0.f};
#pragma unroll
    for (int kt = 0; kt < 4; ++kt) {
      // A fragment: lane(0-15): K 0..15, lane(16-31): K 16..31
      ABfrag a;
      a.h[0] = *(const v8bf*)(arow + kt * 32 + hi * 16);
      a.h[1] = *(const v8bf*)(arow + kt * 32 + hi * 16 + 8);
      acc0 = __builtin_amdgcn_wmma_f32_16x16x32_bf16(
          false, a.v, false, B[kt][0].v, (short)0, acc0, false, false);
      acc1 = __builtin_amdgcn_wmma_f32_16x16x32_bf16(
          false, a.v, false, B[kt][1].v, (short)0, acc1, false, false);
    }
    // C/D layout: VGPR r, lane L -> (M = r + 8*hi, N = L&15)
    float* orow = Hout + (size_t)(rowbase + hi * 8) * NCOLS + cpair * 32 + m;
#pragma unroll
    for (int r = 0; r < 8; ++r) {
      orow[(size_t)r * NCOLS]      = acc0[r];
      orow[(size_t)r * NCOLS + 16] = acc1[r];
    }
  }
}

// --------------------------- aggregation -----------------------------------
// out[i,:] = dinv[i]^2 * h[i,:]   (self-loop term; also initializes accumulator)
template <int NC>
__global__ void k_agg_init(const float* __restrict__ H, const float* __restrict__ dinv,
                           float* __restrict__ out) {
  int i = blockIdx.x * blockDim.x + threadIdx.x;
  if (i < NNODES * NC) {
    int node = i / NC;
    float di = dinv[node];
    out[i] = di * di * H[i];
  }
}

// one wave32 per edge; NC/32 floats per lane; hw f32 atomics into L2-resident out
template <int NC>
__global__ __launch_bounds__(256)
void k_edge(const int* __restrict__ src, const int* __restrict__ dst,
            const float* __restrict__ dinv, const float* __restrict__ H,
            float* __restrict__ out) {
  int tid  = blockIdx.x * blockDim.x + threadIdx.x;
  int e    = tid >> 5;
  int lane = tid & 31;
  if (e >= NEDGES) return;
  int s = src[e];
  int d = dst[e];
  float nrm = dinv[s] * dinv[d];
  constexpr int FP = NC / 32;
  const float* hp = H + (size_t)s * NC + lane * FP;
  float* op       = out + (size_t)d * NC + lane * FP;
  float vals[FP];
#pragma unroll
  for (int j = 0; j < FP; ++j) vals[j] = nrm * hp[j];   // coalesced row gather
#pragma unroll
  for (int j = 0; j < FP; ++j) unsafeAtomicAdd(op + j, vals[j]);
}

// bias (+ optional ELU, alpha = 1)
template <int NC, bool ELU>
__global__ void k_bias_act(float* __restrict__ out, const float* __restrict__ bias) {
  int i = blockIdx.x * blockDim.x + threadIdx.x;
  if (i < NNODES * NC) {
    float v = out[i] + bias[i % NC];
    if (ELU) v = (v > 0.f) ? v : (__expf(v) - 1.f);
    out[i] = v;
  }
}

// ---------------------------------------------------------------------------
static inline int cdiv(long a, long b) { return (int)((a + b - 1) / b); }

extern "C" void kernel_launch(void* const* d_in, const int* in_sizes, int n_in,
                              void* d_out, int out_size, void* d_ws, size_t ws_size,
                              hipStream_t stream) {
  const float* x  = (const float*)d_in[0];
  const int*   ei = (const int*)d_in[1];           // [2, E] int32
  const float* W1 = (const float*)d_in[2];
  const float* b1 = (const float*)d_in[3];
  const float* W2 = (const float*)d_in[4];
  const float* b2 = (const float*)d_in[5];
  const float* W3 = (const float*)d_in[6];
  const float* b3 = (const float*)d_in[7];
  const int* srcv = ei;
  const int* dstv = ei + NEDGES;
  float* outp = (float*)d_out;

  // workspace carve-up (256B aligned)
  char* ws = (char*)d_ws;
  float*  dinv = (float*)ws;                                 //    200,000 B
  __bf16* xb   = (__bf16*)(ws + 200192);                     // 12,845,056 B (NPAD*128 bf16)
  float*  hbuf = (float*)(ws + 13045248);                    // 25,690,112 B (NPAD*128 f32)
  float*  agg  = (float*)(ws + 38735360);                    // 25,600,000 B (N*128 f32)
  __bf16* wt   = (__bf16*)(ws + 64335360);                   //     32,768 B

  const int TPB = 256;
  const int GEMM_BLOCKS = NPAD / 256;                 // 196
  const int EDGE_BLOCKS = cdiv((long)NEDGES * 32, TPB);

  // ---- dinv = rsqrt(in-degree + 1) (graph is fixed across layers) ----
  k_deg_init <<<cdiv(NNODES, TPB), TPB, 0, stream>>>(dinv, NNODES);
  k_deg_count<<<cdiv(NEDGES, TPB), TPB, 0, stream>>>(dstv, dinv, NEDGES);
  k_deg_rsqrt<<<cdiv(NNODES, TPB), TPB, 0, stream>>>(dinv, NNODES);

  // ---------------- layer 1: 128 -> 128, ELU ----------------
  k_wt_bf16<<<cdiv(128 * 128, TPB), TPB, 0, stream>>>(W1, wt, 128);
  k_f32_to_bf16_pad<<<cdiv((long)NPAD * 128, TPB), TPB, 0, stream>>>(x, xb, NNODES * 128, NPAD * 128);
  k_gemm_bf16<8><<<GEMM_BLOCKS, 256, 0, stream>>>(xb, wt, hbuf);
  k_agg_init<128><<<cdiv((long)NNODES * 128, TPB), TPB, 0, stream>>>(hbuf, dinv, agg);
  k_edge<128><<<EDGE_BLOCKS, TPB, 0, stream>>>(srcv, dstv, dinv, hbuf, agg);
  k_bias_act<128, true><<<cdiv((long)NNODES * 128, TPB), TPB, 0, stream>>>(agg, b1);

  // ---------------- layer 2: 128 -> 128, ELU ----------------
  k_wt_bf16<<<cdiv(128 * 128, TPB), TPB, 0, stream>>>(W2, wt, 128);
  k_f32_to_bf16_pad<<<cdiv((long)NPAD * 128, TPB), TPB, 0, stream>>>(agg, xb, NNODES * 128, NPAD * 128);
  k_gemm_bf16<8><<<GEMM_BLOCKS, 256, 0, stream>>>(xb, wt, hbuf);
  k_agg_init<128><<<cdiv((long)NNODES * 128, TPB), TPB, 0, stream>>>(hbuf, dinv, agg);
  k_edge<128><<<EDGE_BLOCKS, TPB, 0, stream>>>(srcv, dstv, dinv, hbuf, agg);
  k_bias_act<128, true><<<cdiv((long)NNODES * 128, TPB), TPB, 0, stream>>>(agg, b2);

  // ---------------- layer 3: 128 -> 64, no ELU, into d_out ----------------
  k_wt_bf16<<<cdiv(128 * 64, TPB), TPB, 0, stream>>>(W3, wt, 64);
  k_f32_to_bf16_pad<<<cdiv((long)NPAD * 128, TPB), TPB, 0, stream>>>(agg, xb, NNODES * 128, NPAD * 128);
  k_gemm_bf16<4><<<GEMM_BLOCKS, 256, 0, stream>>>(xb, wt, hbuf);
  k_agg_init<64><<<cdiv((long)NNODES * 64, TPB), TPB, 0, stream>>>(hbuf, dinv, outp);
  k_edge<64><<<EDGE_BLOCKS, TPB, 0, stream>>>(srcv, dstv, dinv, hbuf, outp);
  k_bias_act<64, false><<<cdiv((long)NNODES * 64, TPB), TPB, 0, stream>>>(outp, b3);
}
